// W8A16LinearStatic_81166291960407
// MI455X (gfx1250) — matrix operand order repaired and checked by reading, not compiled, not run on hardware
//
#include <hip/hip_runtime.h>
#include <stdint.h>

typedef __attribute__((ext_vector_type(16))) _Float16 v16h;
typedef __attribute__((ext_vector_type(8)))  _Float16 v8h;
typedef __attribute__((ext_vector_type(2)))  _Float16 v2h;
typedef __attribute__((ext_vector_type(8)))  float    v8f;

namespace {
constexpr int MDIM = 8192;      // B * S
constexpr int NDIM = 16384;     // D_OUT
constexpr int KDIM = 4096;      // D_IN
constexpr int BM   = 128;
constexpr int BN   = 256;
constexpr int BK   = 32;        // one WMMA k-step per stage
constexpr int LDA  = BK + 8;    // 40 halves = 80 B LDS row stride (16-B aligned)
constexpr int KT   = KDIM / BK; // 128 k-tiles
}

// Dequantize 4 int8 -> 4 fp16 = (fp16(b) + off) * scl, bit-exact vs reference.
// Magic: 0x6400 | (b ^ 0x80) is the exact fp16 encoding of (b + 1152);
// (h - 1152) is exact, so rounding matches ((_Float16)b + off) * scl.
__device__ __forceinline__ uint2 dq4(unsigned int w, v2h off2, v2h scl2) {
    const unsigned int cst = 0x64646464u;
    unsigned int p0 = __builtin_amdgcn_perm(cst, w, 0x05010400u) ^ 0x00800080u;
    unsigned int p1 = __builtin_amdgcn_perm(cst, w, 0x07030602u) ^ 0x00800080u;
    v2h h0 = __builtin_bit_cast(v2h, p0);
    v2h h1 = __builtin_bit_cast(v2h, p1);
    const v2h bias = {(_Float16)1152.0f, (_Float16)1152.0f};
    h0 = ((h0 - bias) + off2) * scl2;   // v_pk_add_f16 x2, v_pk_mul_f16
    h1 = ((h1 - bias) + off2) * scl2;
    uint2 r;
    r.x = __builtin_bit_cast(unsigned int, h0);
    r.y = __builtin_bit_cast(unsigned int, h1);
    return r;
}

// Dequantize 16 int8 (one uint4) into two LDS-ready uint4 (16 halves).
__device__ __forceinline__ void dq16(uint4 w, v2h off2, v2h scl2,
                                     uint4& lo, uint4& hi) {
    uint2 q0 = dq4(w.x, off2, scl2);
    uint2 q1 = dq4(w.y, off2, scl2);
    uint2 q2 = dq4(w.z, off2, scl2);
    uint2 q3 = dq4(w.w, off2, scl2);
    lo.x = q0.x; lo.y = q0.y; lo.z = q1.x; lo.w = q1.y;
    hi.x = q2.x; hi.y = q2.y; hi.z = q3.x; hi.w = q3.y;
}

__global__ __launch_bounds__(256)
void w8a16_wmma_kernel(const _Float16* __restrict__ X,
                       const int8_t*   __restrict__ W,
                       const _Float16* __restrict__ Wscale,
                       const _Float16* __restrict__ Woff,
                       _Float16*       __restrict__ Y)
{
    __shared__ _Float16 As[2][BM * LDA];   // 2 x 10 KB
    __shared__ _Float16 Bs[2][BN * LDA];   // 2 x 20 KB

    const int tid  = threadIdx.x;
    const int lane = tid & 31;
    const int wave = tid >> 5;
    const int wr   = wave >> 2;    // 0..1 : 64-row strip
    const int wc   = wave & 3;     // 0..3 : 64-col strip
    const int hsel = lane >> 4;    // WMMA high-half select
    const int nl   = lane & 15;

    const int bm = blockIdx.y * BM;
    const int bn = blockIdx.x * BN;

    // A staging: 128 rows x 64 B = 8 KB -> 2 passes of 16 B/thread
    const int a_r = tid >> 2;        // + p*64
    const int a_c = (tid & 3) * 8;   // half offset
    // B staging: 256 rows x 32 B = 8 KB -> 2 passes of 16 int8/thread
    const int b_r = tid >> 1;        // + p*128
    const int b_c = (tid & 1) * 16;  // bytes == halves offset (coincide)

    // per-thread dequant constants (splatted to packed-f16 pairs)
    v2h scl2[2], off2[2];
    #pragma unroll
    for (int p = 0; p < 2; ++p) {
        const int n = bn + p * 128 + b_r;
        const _Float16 s = Wscale[n];
        const _Float16 o = Woff[n];
        scl2[p] = (v2h){s, s};
        off2[p] = (v2h){o, o};
    }

    v8f acc[4][4];
    #pragma unroll
    for (int i = 0; i < 4; ++i)
        #pragma unroll
        for (int j = 0; j < 4; ++j)
            #pragma unroll
            for (int e = 0; e < 8; ++e)
                acc[i][j][e] = 0.0f;

    // ---- prologue: stage k-tile 0 into buffer 0 ----
    #pragma unroll
    for (int p = 0; p < 2; ++p) {
        const int r = p * 64 + a_r;
        *(uint4*)(&As[0][r * LDA + a_c]) =
            *(const uint4*)(X + (size_t)(bm + r) * KDIM + a_c);
    }
    #pragma unroll
    for (int p = 0; p < 2; ++p) {
        const int r = p * 128 + b_r;
        uint4 w = *(const uint4*)(W + (size_t)(bn + r) * KDIM + b_c);
        uint4 lo, hi;
        dq16(w, off2[p], scl2[p], lo, hi);
        *(uint4*)(&Bs[0][r * LDA + b_c])     = lo;
        *(uint4*)(&Bs[0][r * LDA + b_c + 8]) = hi;
    }
    __syncthreads();

    for (int kt = 0; kt < KT; ++kt) {
        const int cur = kt & 1;
        const int nxt = cur ^ 1;
        const int k1  = (kt + 1) * BK;

        // ---- issue next k-tile global loads early (hide behind WMMAs) ----
        uint4 a_stage[2], b_stage[2];
        if (kt + 1 < KT) {
            #pragma unroll
            for (int p = 0; p < 2; ++p) {
                const int r = p * 64 + a_r;
                a_stage[p] =
                    *(const uint4*)(X + (size_t)(bm + r) * KDIM + k1 + a_c);
            }
            #pragma unroll
            for (int p = 0; p < 2; ++p) {
                const int r = p * 128 + b_r;
                b_stage[p] =
                    *(const uint4*)(W + (size_t)(bn + r) * KDIM + k1 + b_c);
            }
        }

        // ---- compute on current buffer: 16 WMMAs per wave ----
        v16h afrag[4];
        #pragma unroll
        for (int im = 0; im < 4; ++im) {
            // A fragment: row m, K-chunks [8h,8h+8) and [16+8h,16+8h+8)
            const int m = wr * 64 + im * 16 + nl;
            v8h lo = *(const v8h*)(&As[cur][m * LDA + 8 * hsel]);
            v8h hi = *(const v8h*)(&As[cur][m * LDA + 16 + 8 * hsel]);
            afrag[im] = __builtin_shufflevector(
                lo, hi, 0,1,2,3,4,5,6,7,8,9,10,11,12,13,14,15);
        }
        #pragma unroll
        for (int jn = 0; jn < 4; ++jn) {
            // B fragment: column n = weight row n, K-chunks [8h,8h+8) and [16+8h,16+8h+8)
            const int n = wc * 64 + jn * 16 + nl;
            v8h lo = *(const v8h*)(&Bs[cur][n * LDA + 8 * hsel]);
            v8h hi = *(const v8h*)(&Bs[cur][n * LDA + 16 + 8 * hsel]);
            v16h bfrag = __builtin_shufflevector(
                lo, hi, 0,1,2,3,4,5,6,7,8,9,10,11,12,13,14,15);
            #pragma unroll
            for (int im = 0; im < 4; ++im) {
                acc[im][jn] = __builtin_amdgcn_wmma_f32_16x16x32_f16(
                    /*neg_a=*/false, afrag[im],
                    /*neg_b=*/false, bfrag,
                    /*c_mod=*/(short)0, acc[im][jn],
                    /*reuse_a=*/false, /*reuse_b=*/false);
            }
        }

        // ---- dequant + store staged data into the other buffer ----
        if (kt + 1 < KT) {
            #pragma unroll
            for (int p = 0; p < 2; ++p) {
                const int r = p * 64 + a_r;
                *(uint4*)(&As[nxt][r * LDA + a_c]) = a_stage[p];
            }
            #pragma unroll
            for (int p = 0; p < 2; ++p) {
                const int r = p * 128 + b_r;
                uint4 lo, hi;
                dq16(b_stage[p], off2[p], scl2[p], lo, hi);
                *(uint4*)(&Bs[nxt][r * LDA + b_c])     = lo;
                *(uint4*)(&Bs[nxt][r * LDA + b_c + 8]) = hi;
            }
        }
        __syncthreads();
    }

    // ---- epilogue: f32 -> fp16, nontemporal (y is streaming, no reuse) ----
    // C/D layout: VGPR r holds row (r + 8*hsel), column nl
    #pragma unroll
    for (int im = 0; im < 4; ++im) {
        #pragma unroll
        for (int jn = 0; jn < 4; ++jn) {
            const int col  = bn + wc * 64 + jn * 16 + nl;
            const int row0 = bm + wr * 64 + im * 16 + 8 * hsel;
            #pragma unroll
            for (int r = 0; r < 8; ++r) {
                __builtin_nontemporal_store(
                    (_Float16)acc[im][jn][r],
                    &Y[(size_t)(row0 + r) * NDIM + col]);
            }
        }
    }
}

extern "C" void kernel_launch(void* const* d_in, const int* in_sizes, int n_in,
                              void* d_out, int out_size, void* d_ws, size_t ws_size,
                              hipStream_t stream) {
    const _Float16* X  = (const _Float16*)d_in[0];   // [4,2048,4096] fp16
    const int8_t*   W  = (const int8_t*)d_in[1];     // [16384,4096] int8
    const _Float16* Ws = (const _Float16*)d_in[2];   // [16384] fp16 scale
    const _Float16* Wo = (const _Float16*)d_in[3];   // [16384] fp16 (negated) offset
    _Float16* Y = (_Float16*)d_out;                  // [4,2048,16384] fp16

    dim3 grid(NDIM / BN, MDIM / BM);                 // 64 x 64 blocks
    w8a16_wmma_kernel<<<grid, dim3(256), 0, stream>>>(X, W, Ws, Wo, Y);
}
